// ActorCriticRNN_21663815041201
// MI455X (gfx1250) — matrix operand order repaired and checked by reading, not compiled, not run on hardware
//
#include <hip/hip_runtime.h>
#include <hip/hip_bf16.h>

// ---------------- problem constants ----------------
#define T_   256
#define B_   512
#define OBS_ 256
#define FC_  512
#define H_   512
#define A_   16
#define H3_  1536
#define MB_  (T_*B_)          // 131072 rows for the batched GEMMs

typedef __attribute__((ext_vector_type(16))) __bf16 v16bf;
typedef __attribute__((ext_vector_type(8)))  float  v8f;

__device__ __forceinline__ unsigned short f2bf(float f) {
    unsigned u = __float_as_uint(f);
    unsigned r = (u + 0x7fffu + ((u >> 16) & 1u)) >> 16;   // RNE
    return (unsigned short)r;
}
__device__ __forceinline__ float bf2f(unsigned short h) {
    return __uint_as_float(((unsigned)h) << 16);
}
__device__ __forceinline__ float sigmoidf_(float x) { return 1.0f / (1.0f + __expf(-x)); }
// A-fragment k offset for dword i (0..7), lane group g (0..1)  [contiguous: k = 16g + 2i, matches packed B]
__device__ __forceinline__ int a_kbase(int i, int g) {
    return g * 16 + 2 * i;
}
// CDNA5 async copy: 16 bytes global -> LDS, tracked by ASYNCcnt (ISA 10.x / 15.18.3 op 98)
__device__ __forceinline__ void async_cp16(unsigned lds_off, const void* gptr) {
    asm volatile("global_load_async_to_lds_b128 %0, %1, off"
                 :: "v"(lds_off), "v"((unsigned long long)gptr) : "memory");
}
__device__ __forceinline__ void wait_async0() {
    asm volatile("s_wait_asynccnt 0" ::: "memory");
}

// ---------------- elementwise converts / packing ----------------
__global__ void cvt_f32_to_bf16(const float* __restrict__ x, unsigned short* __restrict__ o, long n) {
    long i = (long)blockIdx.x * blockDim.x + threadIdx.x;
    if (i < n) o[i] = f2bf(x[i]);
}

// Pack weight W[Ksrc,Nsrc] (row-major, K = input dim) into fragment-major bf16 tiles.
// Tile (kt,nt) = 512 bf16: lane L (0..31) holds column n = nt*16 + L%16,
// j (0..15) -> k = kt*32 + (L/16)*16 + j.   Zero-pad columns >= Nsrc.
__global__ void pack_weight(const float* __restrict__ W, unsigned short* __restrict__ pk,
                            int Ksrc, int Nsrc, int Np) {
    long idx = (long)blockIdx.x * blockDim.x + threadIdx.x;
    long total = (long)Ksrc * Np;
    if (idx >= total) return;
    long tile = idx >> 9;
    int  r    = (int)(idx & 511);
    int  L = r >> 4, j = r & 15;
    int  ntiles = Np >> 4;
    int  kt = (int)(tile / ntiles), nt = (int)(tile % ntiles);
    int  k = kt * 32 + (L >> 4) * 16 + j;
    int  n = nt * 16 + (L & 15);
    float v = (n < Nsrc) ? W[(long)k * Nsrc + n] : 0.0f;
    pk[idx] = f2bf(v);
}

// ---------------- big GEMM: C = act(A_bf16[M,K] x Bpk[K,N] + bias) ----------------
// grid = (N/64, M/128), block = 256 (8 waves). Wave w: rows m0+16w..+15, 4 n-tiles.
// A is staged 128x32 bf16 per K-step via async global->LDS copies, double buffered:
//   issue(0); for kt: wait_async -> barrier -> issue(kt+1, buf^1) -> wmma(buf)
__global__ __launch_bounds__(256) void gemm_bf16(
        const unsigned short* __restrict__ A, const unsigned short* __restrict__ Bpk,
        const float* __restrict__ bias, float* __restrict__ Cf, unsigned short* __restrict__ Cb,
        int K, int N, int relu) {
    __shared__ unsigned short sA[2 * 128 * 32];   // 2 x 8 KB double buffer
    const int tid = threadIdx.x;
    const int w = tid >> 5, lane = tid & 31, lm = lane & 15, g = lane >> 4;
    const long m0 = (long)blockIdx.y * 128;
    const int  n0 = blockIdx.x * 64;
    const int  ntiles = N >> 4;
    const int  nk = K >> 5;

    const unsigned sA_base = (unsigned)(unsigned long long)(const void*)sA;
    const int crow = tid >> 1, chalf = tid & 1;            // 32 B per thread per stage
    const unsigned short* gsrc0 = A + (m0 + crow) * K + chalf * 16;

    v8f acc[4];
    #pragma unroll
    for (int i = 0; i < 4; ++i)
        #pragma unroll
        for (int q = 0; q < 8; ++q) acc[i][q] = 0.0f;

    // prologue: stage kt = 0 into buffer 0
    {
        unsigned l0 = sA_base + crow * 64 + chalf * 32;
        async_cp16(l0,      gsrc0);
        async_cp16(l0 + 16, gsrc0 + 8);
    }

    for (int kt = 0; kt < nk; ++kt) {
        wait_async0();          // this wave's pending async copies have landed
        __syncthreads();        // all waves landed; prior reads of buf^1 complete
        if (kt + 1 < nk) {      // overlap next stage with this step's WMMAs
            int nb = (kt + 1) & 1;
            unsigned l0 = sA_base + nb * 8192 + crow * 64 + chalf * 32;
            const unsigned short* gp = gsrc0 + (kt + 1) * 32;
            async_cp16(l0,      gp);
            async_cp16(l0 + 16, gp + 8);
        }
        const unsigned short* buf = sA + (kt & 1) * 4096;
        union { v16bf v; unsigned u[8]; } af;
        const int rowbase = (w * 16 + lm) * 32;
        #pragma unroll
        for (int i = 0; i < 8; ++i)
            af.u[i] = *reinterpret_cast<const unsigned*>(buf + rowbase + a_kbase(i, g));
        #pragma unroll
        for (int nt = 0; nt < 4; ++nt) {
            v16bf bf = *reinterpret_cast<const v16bf*>(
                Bpk + ((long)kt * ntiles + (n0 >> 4) + nt) * 512 + lane * 16);
            acc[nt] = __builtin_amdgcn_wmma_f32_16x16x32_bf16(
                false, af.v, false, bf, (short)0, acc[nt], false, false);
        }
    }
    #pragma unroll
    for (int nt = 0; nt < 4; ++nt) {
        int col = n0 + nt * 16 + lm;
        float bsv = bias ? bias[col] : 0.0f;
        #pragma unroll
        for (int v = 0; v < 8; ++v) {
            long row = m0 + w * 16 + v + 8 * g;
            float x = acc[nt][v] + bsv;
            if (relu) x = x > 0.0f ? x : 0.0f;
            if (Cf) Cf[row * N + col] = x;
            if (Cb) Cb[row * N + col] = f2bf(x);
        }
    }
}

// ---------------- narrow GEMM (N = 16 tile) for head outputs ----------------
// grid = M/128, block = 256. Writes out[row*ostride + col] for col < nvalid.
__global__ __launch_bounds__(256) void gemm_narrow(
        const unsigned short* __restrict__ A, const unsigned short* __restrict__ Bpk,
        const float* __restrict__ bias, float* __restrict__ outp,
        int K, int nvalid, int ostride) {
    const int tid = threadIdx.x;
    const int w = tid >> 5, lane = tid & 31, lm = lane & 15, g = lane >> 4;
    const long m0 = (long)blockIdx.x * 128 + w * 16;
    v8f acc;
    #pragma unroll
    for (int q = 0; q < 8; ++q) acc[q] = 0.0f;
    for (int kt = 0; kt < (K >> 5); ++kt) {
        union { v16bf v; unsigned u[8]; } af;
        const long rowoff = (m0 + lm) * K + kt * 32;
        #pragma unroll
        for (int i = 0; i < 8; ++i)
            af.u[i] = *reinterpret_cast<const unsigned*>(A + rowoff + a_kbase(i, g));
        v16bf bf = *reinterpret_cast<const v16bf*>(Bpk + (long)kt * 512 + lane * 16);
        acc = __builtin_amdgcn_wmma_f32_16x16x32_bf16(false, af.v, false, bf, (short)0, acc, false, false);
    }
    int col = lm;
    if (col < nvalid) {
        float bsv = bias ? bias[col] : 0.0f;
        #pragma unroll
        for (int v = 0; v < 8; ++v) {
            long row = m0 + v + 8 * g;
            outp[row * ostride + col] = acc[v] + bsv;
        }
    }
}

// ---------------- GRU step: one timestep, block = 16 batch rows x all 1536 cols ----------------
// block = 512 threads (16 waves); wave w owns n-tiles {w+16*i}: i=p -> r, i=p+2 -> z, i=p+4 -> n
// for output unit j = p*256 + 16*w + lane%16, p in {0,1}.
__global__ __launch_bounds__(512) void gru_step(
        const float* __restrict__ hprev, const int* __restrict__ dones_t,
        const float* __restrict__ xi_t, const unsigned short* __restrict__ Whpk,
        const float* __restrict__ bhn,
        float* __restrict__ yout, unsigned short* __restrict__ yb) {
    __shared__ unsigned short sB16[16 * 512];  // masked h, bf16 (A matrix)
    __shared__ float          sF[16 * 512];    // masked h, fp32 (for blend)
    const int tid = threadIdx.x;
    const int b0 = blockIdx.x * 16;

    #pragma unroll
    for (int i = 0; i < 16; ++i) {
        float hv = hprev[(long)(b0 + i) * H_ + tid];
        if (dones_t[b0 + i]) hv = 0.0f;
        sF[i * 512 + tid] = hv;
        sB16[i * 512 + tid] = f2bf(hv);
    }
    __syncthreads();

    const int w = tid >> 5, lane = tid & 31, lm = lane & 15, g = lane >> 4;
    v8f acc[6];
    #pragma unroll
    for (int i = 0; i < 6; ++i)
        #pragma unroll
        for (int q = 0; q < 8; ++q) acc[i][q] = 0.0f;

    for (int kt = 0; kt < 16; ++kt) {              // K = 512
        union { v16bf v; unsigned u[8]; } af;
        const int rowbase = lm * 512 + kt * 32;
        #pragma unroll
        for (int i = 0; i < 8; ++i)
            af.u[i] = *reinterpret_cast<const unsigned*>(sB16 + rowbase + a_kbase(i, g));
        #pragma unroll
        for (int i = 0; i < 6; ++i) {
            int nt = w + 16 * i;
            v16bf bf = *reinterpret_cast<const v16bf*>(Whpk + ((long)kt * 96 + nt) * 512 + lane * 16);
            acc[i] = __builtin_amdgcn_wmma_f32_16x16x32_bf16(
                false, af.v, false, bf, (short)0, acc[i], false, false);
        }
    }
    #pragma unroll
    for (int p = 0; p < 2; ++p) {
        int j = p * 256 + w * 16 + lm;
        float bh = bhn[j];
        #pragma unroll
        for (int v = 0; v < 8; ++v) {
            int brow = v + 8 * g;
            long base = (long)(b0 + brow) * H3_;
            float r  = sigmoidf_(xi_t[base + j]          + acc[p][v]);
            float z  = sigmoidf_(xi_t[base + 512 + j]    + acc[p + 2][v]);
            float nn = tanhf   (xi_t[base + 1024 + j]    + r * (acc[p + 4][v] + bh));
            float h0 = sF[brow * 512 + j];
            float hn = (1.0f - z) * nn + z * h0;
            long oidx = (long)(b0 + brow) * H_ + j;
            yout[oidx] = hn;
            yb[oidx]   = f2bf(hn);
        }
    }
}

// ---------------- dormancy reductions ----------------
// per-(b,j) sum over T of |x[t,b,j]|; x is [T, BN] flat; one thread per (b,j)
__global__ void colsum_T(const void* __restrict__ X, int isbf16, float* __restrict__ dsum, int BN) {
    int idx = blockIdx.x * blockDim.x + threadIdx.x;
    if (idx >= BN) return;
    float s = 0.0f;
    if (isbf16) {
        const unsigned short* xb = (const unsigned short*)X;
        for (int t = 0; t < T_; ++t) s += fabsf(bf2f(xb[(long)t * BN + idx]));
    } else {
        const float* xf = (const float*)X;
        for (int t = 0; t < T_; ++t) s += fabsf(xf[(long)t * BN + idx]);
    }
    dsum[idx] = s;
}

__global__ void colsum_hidden(const float* __restrict__ hfin, float* __restrict__ dsum) {
    int j = blockIdx.x * blockDim.x + threadIdx.x;
    if (j >= H_) return;
    float s = 0.0f;
    for (int b = 0; b < B_; ++b) s += fabsf(hfin[(long)b * H_ + j]);
    dsum[j] = s;
}

// single block: d = dsum/nrows; denom = sum(d)/layer_dim + 1e-8; pct of (d/denom <= 0)
__global__ __launch_bounds__(1024) void dorm_final(const float* __restrict__ dsum, int nelem,
                                                   float nrows, float layer_dim, float* __restrict__ outp) {
    __shared__ float red[1024];
    float s = 0.0f;
    for (int i = threadIdx.x; i < nelem; i += 1024) s += dsum[i];
    red[threadIdx.x] = s; __syncthreads();
    for (int o = 512; o > 0; o >>= 1) {
        if (threadIdx.x < o) red[threadIdx.x] += red[threadIdx.x + o];
        __syncthreads();
    }
    float denom = red[0] / nrows / layer_dim + 1e-8f;
    __syncthreads();
    float c = 0.0f;
    for (int i = threadIdx.x; i < nelem; i += 1024) {
        float d = dsum[i] / nrows;
        if (d / denom <= 0.0f) c += 1.0f;
    }
    red[threadIdx.x] = c; __syncthreads();
    for (int o = 512; o > 0; o >>= 1) {
        if (threadIdx.x < o) red[threadIdx.x] += red[threadIdx.x + o];
        __syncthreads();
    }
    if (threadIdx.x == 0) *outp = red[0] / layer_dim * 100.0f;
}

__global__ void sigma_kernel(const float* __restrict__ log_std, float* __restrict__ o) {
    int i = threadIdx.x;
    if (i < A_) o[i] = __expf(log_std[i]);
}

// ---------------- host launcher ----------------
extern "C" void kernel_launch(void* const* d_in, const int* in_sizes, int n_in,
                              void* d_out, int out_size, void* d_ws, size_t ws_size,
                              hipStream_t stream) {
    const float* hidden    = (const float*)d_in[0];
    const float* obs       = (const float*)d_in[1];
    const int*   dones     = (const int*)d_in[2];
    const float* W_emb     = (const float*)d_in[3];
    const float* b_emb     = (const float*)d_in[4];
    const float* Wi        = (const float*)d_in[5];
    const float* bi        = (const float*)d_in[6];
    const float* Wh        = (const float*)d_in[7];
    const float* bhn       = (const float*)d_in[8];
    const float* W_act     = (const float*)d_in[9];
    const float* b_act     = (const float*)d_in[10];
    const float* W_act_out = (const float*)d_in[11];
    const float* b_act_out = (const float*)d_in[12];
    const float* log_std   = (const float*)d_in[13];
    const float* W_cr      = (const float*)d_in[14];
    const float* b_cr      = (const float*)d_in[15];
    const float* W_cr_out  = (const float*)d_in[16];
    const float* b_cr_out  = (const float*)d_in[17];
    (void)in_sizes; (void)n_in; (void)out_size; (void)ws_size;

    float* out = (float*)d_out;
    float* out_hidden = out;                                  // [B,H]
    float* out_amean  = out + (long)B_ * H_;                  // [T,B,A]
    float* out_sigma  = out_amean + (long)MB_ * A_;           // [A]
    float* out_value  = out_sigma + A_;                       // [T,B]
    float* out_dorm   = out_value + (long)MB_;                // 5 scalars: ad1,ed1,hd1,ed2,cd1

    // ---- carve workspace ----
    char* p = (char*)d_ws;
    auto alloc = [&](size_t bytes) -> void* {
        void* r = (void*)p; p += (bytes + 255) & ~(size_t)255; return r;
    };
    unsigned short* obs_b   = (unsigned short*)alloc((size_t)MB_ * OBS_ * 2);
    unsigned short* emb_b   = (unsigned short*)alloc((size_t)MB_ * FC_ * 2);
    unsigned short* y_b     = (unsigned short*)alloc((size_t)MB_ * H_ * 2);
    unsigned short* act_b   = (unsigned short*)alloc((size_t)MB_ * FC_ * 2);
    unsigned short* cr_b    = (unsigned short*)alloc((size_t)MB_ * FC_ * 2);
    unsigned short* Wemb_pk = (unsigned short*)alloc((size_t)OBS_ * FC_ * 2);
    unsigned short* Wi_pk   = (unsigned short*)alloc((size_t)FC_ * H3_ * 2);
    unsigned short* Wh_pk   = (unsigned short*)alloc((size_t)H_ * H3_ * 2);
    unsigned short* Wact_pk = (unsigned short*)alloc((size_t)H_ * FC_ * 2);
    unsigned short* Wcr_pk  = (unsigned short*)alloc((size_t)H_ * FC_ * 2);
    unsigned short* Wao_pk  = (unsigned short*)alloc((size_t)FC_ * 16 * 2);
    unsigned short* Wco_pk  = (unsigned short*)alloc((size_t)FC_ * 16 * 2);
    float* xi   = (float*)alloc((size_t)MB_ * H3_ * 4);
    float* yf   = (float*)alloc((size_t)MB_ * H_ * 4);
    float* ds_e = (float*)alloc((size_t)B_ * FC_ * 4);
    float* ds_y = (float*)alloc((size_t)B_ * H_ * 4);
    float* ds_a = (float*)alloc((size_t)B_ * FC_ * 4);
    float* ds_c = (float*)alloc((size_t)B_ * FC_ * 4);
    float* ds_h = (float*)alloc((size_t)H_ * 4);

    const int BN = B_ * 512;

    // ---- pack weights to fragment-major bf16 ----
    auto packN = [&](const float* W, unsigned short* pk, int K, int Nsrc, int Np) {
        long total = (long)K * Np;
        pack_weight<<<dim3((unsigned)((total + 255) / 256)), dim3(256), 0, stream>>>(W, pk, K, Nsrc, Np);
    };
    packN(W_emb,     Wemb_pk, OBS_, FC_,  FC_);
    packN(Wi,        Wi_pk,   FC_,  H3_,  H3_);
    packN(Wh,        Wh_pk,   H_,   H3_,  H3_);
    packN(W_act,     Wact_pk, H_,   FC_,  FC_);
    packN(W_cr,      Wcr_pk,  H_,   FC_,  FC_);
    packN(W_act_out, Wao_pk,  FC_,  16,   16);
    packN(W_cr_out,  Wco_pk,  FC_,  1,    16);

    // ---- obs -> bf16 ----
    {
        long n = (long)MB_ * OBS_;
        cvt_f32_to_bf16<<<dim3((unsigned)((n + 255) / 256)), dim3(256), 0, stream>>>(obs, obs_b, n);
    }

    // ---- emb = relu(obs @ W_emb + b_emb)  [bf16 out] ----
    gemm_bf16<<<dim3(FC_ / 64, MB_ / 128), dim3(256), 0, stream>>>(
        obs_b, Wemb_pk, b_emb, nullptr, emb_b, OBS_, FC_, 1);
    colsum_T<<<dim3(BN / 256), dim3(256), 0, stream>>>(emb_b, 1, ds_e, BN);

    // ---- xi = emb @ Wi + bi  [fp32 out] ----
    gemm_bf16<<<dim3(H3_ / 64, MB_ / 128), dim3(256), 0, stream>>>(
        emb_b, Wi_pk, bi, xi, nullptr, FC_, H3_, 0);

    // ---- GRU scan ----
    for (int t = 0; t < T_; ++t) {
        const float* hp = (t == 0) ? hidden : (yf + (long)(t - 1) * B_ * H_);
        gru_step<<<dim3(B_ / 16), dim3(512), 0, stream>>>(
            hp, dones + (long)t * B_, xi + (long)t * B_ * H3_, Wh_pk, bhn,
            yf + (long)t * B_ * H_, y_b + (long)t * B_ * H_);
    }

    // ---- final hidden out + dormancies of hidden / y ----
    hipMemcpyAsync(out_hidden, yf + (long)(T_ - 1) * B_ * H_, (size_t)B_ * H_ * sizeof(float),
                   hipMemcpyDeviceToDevice, stream);
    colsum_hidden<<<dim3(2), dim3(256), 0, stream>>>(yf + (long)(T_ - 1) * B_ * H_, ds_h);
    colsum_T<<<dim3(BN / 256), dim3(256), 0, stream>>>(yf, 0, ds_y, BN);

    // ---- actor head ----
    gemm_bf16<<<dim3(FC_ / 64, MB_ / 128), dim3(256), 0, stream>>>(
        y_b, Wact_pk, b_act, nullptr, act_b, H_, FC_, 1);
    colsum_T<<<dim3(BN / 256), dim3(256), 0, stream>>>(act_b, 1, ds_a, BN);
    gemm_narrow<<<dim3(MB_ / 128), dim3(256), 0, stream>>>(
        act_b, Wao_pk, b_act_out, out_amean, FC_, 16, 16);

    // ---- critic head ----
    gemm_bf16<<<dim3(FC_ / 64, MB_ / 128), dim3(256), 0, stream>>>(
        y_b, Wcr_pk, b_cr, nullptr, cr_b, H_, FC_, 1);
    colsum_T<<<dim3(BN / 256), dim3(256), 0, stream>>>(cr_b, 1, ds_c, BN);
    gemm_narrow<<<dim3(MB_ / 128), dim3(256), 0, stream>>>(
        cr_b, Wco_pk, b_cr_out, out_value, FC_, 1, 1);

    // ---- sigma ----
    sigma_kernel<<<dim3(1), dim3(32), 0, stream>>>(log_std, out_sigma);

    // ---- dormancy scalars (ad1, ed1, hd1, ed2, cd1) ----
    dorm_final<<<dim3(1), dim3(1024), 0, stream>>>(ds_a, BN,  (float)T_, 512.0f, out_dorm + 0);
    dorm_final<<<dim3(1), dim3(1024), 0, stream>>>(ds_e, BN,  (float)T_, 512.0f, out_dorm + 1);
    dorm_final<<<dim3(1), dim3(1024), 0, stream>>>(ds_h, H_,  (float)B_, 512.0f, out_dorm + 2);
    dorm_final<<<dim3(1), dim3(1024), 0, stream>>>(ds_y, BN,  (float)T_, 512.0f, out_dorm + 3);
    dorm_final<<<dim3(1), dim3(1024), 0, stream>>>(ds_c, BN,  (float)T_, 256.0f, out_dorm + 4);
}
